// MultiHeadAttentionKVCache_11690900979798
// MI455X (gfx1250) — matrix operand order repaired and checked by reading, not compiled, not run on hardware
//
#include <hip/hip_runtime.h>
#include <hip/hip_bf16.h>

// ---------------------------------------------------------------------------
// MI455X (gfx1250) fused MHA w/ KV cache.
// All matmuls via v_wmma_f32_16x16x32_bf16 (f32 accumulate).
// Attention stages K/V tiles in double-buffered LDS via async global->LDS.
// ---------------------------------------------------------------------------

typedef __attribute__((ext_vector_type(8)))  float          v8f;
typedef __attribute__((ext_vector_type(8)))  unsigned short u16x8;
typedef __attribute__((ext_vector_type(16))) __bf16         bf16x16;
typedef __attribute__((ext_vector_type(4)))  int            v4i;

#define D_MODEL 2048
#define NHEAD   16
#define HD      128
#define TQ      1024
#define TP      3072
#define SEQ     4096     // TP + TQ
#define LDSTR   40       // padded LDS row stride (elems) -> 80B, conflict-free
#define KSTR    136      // K tile LDS row stride (elems) -> 272B, conflict-free
#define VSTR    40       // V tile LDS row stride (elems) -> 80B, conflict-free

#if defined(__has_builtin)
#if __has_builtin(__builtin_amdgcn_global_load_async_to_lds_b128)
#define HAVE_ASYNC_LDS 1
#endif
#endif
#ifndef HAVE_ASYNC_LDS
#define HAVE_ASYNC_LDS 0
#endif

// round-to-nearest-even f32 -> bf16 bits
static __device__ __forceinline__ unsigned short f2bf(float f) {
    union { float f; unsigned u; } v; v.f = f;
    unsigned r = v.u + 0x7FFFu + ((v.u >> 16) & 1u);
    return (unsigned short)(r >> 16);
}

static __device__ __forceinline__ bf16x16 mkfrag(u16x8 lo, u16x8 hi) {
    union { u16x8 h[2]; bf16x16 v; } u;
    u.h[0] = lo; u.h[1] = hi;
    return u.v;
}

static __device__ __forceinline__ v8f wmma_bf16(bf16x16 a, bf16x16 b, v8f c) {
    return __builtin_amdgcn_wmma_f32_16x16x32_bf16(
        /*neg_a=*/false, a, /*neg_b=*/false, b,
        /*c_mod=*/(short)0, c, /*reuse_a=*/false, /*reuse_b=*/false);
}

// 16-byte global -> LDS copy; async (ASYNCcnt) when available.
static __device__ __forceinline__ void cp16_g2l(unsigned short* l, const unsigned short* g) {
#if HAVE_ASYNC_LDS
    __builtin_amdgcn_global_load_async_to_lds_b128(
        (__attribute__((address_space(1))) v4i*)(v4i*)const_cast<unsigned short*>(g),
        (__attribute__((address_space(3))) v4i*)(v4i*)l,
        0, 0);
#else
    *(u16x8*)l = *(const u16x8*)g;
#endif
}

// ---------------------------------------------------------------------------
// Kernel 1: copy k_cache/v_cache (fp32) into output K/V regions (rows s<TP)
// and write bf16 copy of K cache.  [B,H,TP,HD] -> [B,H,SEQ,HD]
// ---------------------------------------------------------------------------
__global__ __launch_bounds__(256) void copy_caches(
    const float* __restrict__ kc, const float* __restrict__ vc,
    float* __restrict__ Kout, float* __restrict__ Vout,
    unsigned short* __restrict__ Kb)
{
    size_t idx = (size_t)blockIdx.x * 256 + threadIdx.x;   // one float4 each
    size_t l   = idx * 4;
    const size_t per = (size_t)TP * HD;                    // 393216
    size_t bh  = l / per;
    size_t rem = l - bh * per;
    size_t dst = bh * ((size_t)SEQ * HD) + rem;

    float4 kv = *(const float4*)(kc + l);
    *(float4*)(Kout + dst) = kv;
    unsigned short* kb = Kb + dst;
    kb[0] = f2bf(kv.x); kb[1] = f2bf(kv.y); kb[2] = f2bf(kv.z); kb[3] = f2bf(kv.w);

    float4 vv = *(const float4*)(vc + l);
    *(float4*)(Vout + dst) = vv;
}

// ---------------------------------------------------------------------------
// Kernel 2: v_cache -> bf16 V^T [B,H,HD,SEQ] (cache rows s<TP), 64x64 tiles
// ---------------------------------------------------------------------------
__global__ __launch_bounds__(256) void vt_cache(
    const float* __restrict__ vc, unsigned short* __restrict__ Vt)
{
    __shared__ unsigned short T[64 * 65];
    int tid = threadIdx.x;
    int bidx = blockIdx.x;                 // bh(64) * st(48) * dt(2)
    int dt = bidx & 1;
    int st = (bidx >> 1) % 48;
    int bh = bidx / 96;
    int s0 = st * 64, d0 = dt * 64;

    const float* src = vc + (size_t)bh * TP * HD;
    for (int i = 0; i < 4; ++i) {
        int e4  = tid + i * 256;           // 1024 float4 = 64x64 f32
        int row = e4 >> 4, c4 = e4 & 15;
        float4 v = *(const float4*)(src + (size_t)(s0 + row) * HD + d0 + c4 * 4);
        unsigned short* p = &T[row * 65 + c4 * 4];
        p[0] = f2bf(v.x); p[1] = f2bf(v.y); p[2] = f2bf(v.z); p[3] = f2bf(v.w);
    }
    __syncthreads();
    unsigned short* dst = Vt + (size_t)bh * HD * SEQ;
    for (int i = 0; i < 8; ++i) {
        int e2  = tid + i * 256;           // 2048 ushort2
        int dr  = e2 >> 5, sc2 = e2 & 31;
        unsigned a = T[(sc2 * 2) * 65 + dr];
        unsigned b = T[(sc2 * 2 + 1) * 65 + dr];
        *(unsigned*)(dst + (size_t)(d0 + dr) * SEQ + s0 + sc2 * 2) = a | (b << 16);
    }
}

// ---------------------------------------------------------------------------
// Kernel 3: y = A @ W^T + bias as bf16 WMMA GEMM (TN: both operands K-major).
// Tiles: block 128x128, BK=32, 8 waves, wave = 32x64 = 2x4 WMMA tiles.
// MODE 0: Q  -> bf16 Qb[B,H,TQ,HD]
// MODE 1: K  -> fp32 Kout rows TP.. + bf16 Kb (same layout)
// MODE 2: V  -> fp32 Vout rows TP.. + bf16 Vt (transposed [B,H,HD,SEQ])
// MODE 3: O  -> fp32 out[B*TQ, D]   (A source is bf16 attention output)
// ---------------------------------------------------------------------------
template <int MODE>
__global__ __launch_bounds__(256) void proj_gemm(
    const float* __restrict__ A32, const unsigned short* __restrict__ A16,
    const float* __restrict__ W, const float* __restrict__ bias,
    float* __restrict__ out32, unsigned short* __restrict__ out16)
{
    __shared__ unsigned short As[128 * LDSTR];
    __shared__ unsigned short Bs[128 * LDSTR];

    int tid = threadIdx.x;
    int mt = blockIdx.x >> 4;           // 32 M tiles
    int nt = blockIdx.x & 15;           // 16 N tiles
    int m0 = mt * 128, n0 = nt * 128;
    int w = tid >> 5, lane = tid & 31;
    int wm = w & 3, wn = w >> 2;        // wave -> 32-row group, 64-col group
    int lr = lane & 15, lh = lane >> 4;

    v8f acc[2][4];
    const v8f vz = {0.f,0.f,0.f,0.f,0.f,0.f,0.f,0.f};
    for (int mi = 0; mi < 2; ++mi)
        for (int ni = 0; ni < 4; ++ni) acc[mi][ni] = vz;

    for (int k0 = 0; k0 < D_MODEL; k0 += 32) {
        __syncthreads();
        if (MODE == 3) {
            for (int i = 0; i < 2; ++i) {
                int e8 = tid + i * 256;            // 512 8-elem chunks
                int row = e8 >> 2, c8 = e8 & 3;
                u16x8 v = *(const u16x8*)(A16 + (size_t)(m0 + row) * D_MODEL + k0 + c8 * 8);
                *(u16x8*)(&As[row * LDSTR + c8 * 8]) = v;
                if (k0 + 32 < D_MODEL)
                    __builtin_prefetch(A16 + (size_t)(m0 + row) * D_MODEL + k0 + 32 + c8 * 8, 0, 0);
            }
        } else {
            for (int i = 0; i < 4; ++i) {
                int e4 = tid + i * 256;            // 1024 float4
                int row = e4 >> 3, c4 = e4 & 7;
                float4 v = *(const float4*)(A32 + (size_t)(m0 + row) * D_MODEL + k0 + c4 * 4);
                unsigned short* p = &As[row * LDSTR + c4 * 4];
                p[0] = f2bf(v.x); p[1] = f2bf(v.y); p[2] = f2bf(v.z); p[3] = f2bf(v.w);
                if (k0 + 32 < D_MODEL)
                    __builtin_prefetch(A32 + (size_t)(m0 + row) * D_MODEL + k0 + 32 + c4 * 4, 0, 0);
            }
        }
        for (int i = 0; i < 4; ++i) {              // W rows n0..n0+127 (K-major)
            int e4 = tid + i * 256;
            int row = e4 >> 3, c4 = e4 & 7;
            float4 v = *(const float4*)(W + (size_t)(n0 + row) * D_MODEL + k0 + c4 * 4);
            unsigned short* p = &Bs[row * LDSTR + c4 * 4];
            p[0] = f2bf(v.x); p[1] = f2bf(v.y); p[2] = f2bf(v.z); p[3] = f2bf(v.w);
            if (k0 + 32 < D_MODEL)
                __builtin_prefetch(W + (size_t)(n0 + row) * D_MODEL + k0 + 32 + c4 * 4, 0, 0);
        }
        __syncthreads();

        bf16x16 af[2], bf[4];
        for (int mi = 0; mi < 2; ++mi) {           // A: lane=row, K {0-7,16-23}/{8-15,24-31}
            int row = wm * 32 + mi * 16 + lr;
            int kb = lh * 8;
            u16x8 lo = *(const u16x8*)(&As[row * LDSTR + kb]);
            u16x8 hi = *(const u16x8*)(&As[row * LDSTR + kb + 16]);
            af[mi] = mkfrag(lo, hi);
        }
        for (int ni = 0; ni < 4; ++ni) {           // B: lane=col, K {0-7,16-23}/{8-15,24-31}
            int col = wn * 64 + ni * 16 + lr;
            int kb = lh * 8;
            u16x8 lo = *(const u16x8*)(&Bs[col * LDSTR + kb]);
            u16x8 hi = *(const u16x8*)(&Bs[col * LDSTR + kb + 16]);
            bf[ni] = mkfrag(lo, hi);
        }
        for (int mi = 0; mi < 2; ++mi)
            for (int ni = 0; ni < 4; ++ni)
                acc[mi][ni] = wmma_bf16(af[mi], bf[ni], acc[mi][ni]);
    }

    // epilogue: C layout lane=col(lr), rows j + 8*lh
    for (int mi = 0; mi < 2; ++mi)
        for (int ni = 0; ni < 4; ++ni) {
            int gm = m0 + wm * 32 + mi * 16 + lh * 8;
            int gn = n0 + wn * 64 + ni * 16 + lr;
            float bv = bias[gn];
            v8f a = acc[mi][ni];
            for (int j = 0; j < 8; ++j) {
                float val = a[j] + bv;
                int m = gm + j;
                if (MODE == 3) {
                    out32[(size_t)m * D_MODEL + gn] = val;
                } else {
                    int b = m >> 10, t = m & 1023;
                    int h = gn >> 7, hd = gn & 127;
                    size_t bh = (size_t)(b * NHEAD + h);
                    if (MODE == 0) {
                        out16[(bh * TQ + t) * HD + hd] = f2bf(val);
                    } else {
                        size_t o = (bh * SEQ + TP + t) * HD + hd;
                        out32[o] = val;
                        if (MODE == 1) out16[o] = f2bf(val);
                        else           out16[(bh * HD + hd) * SEQ + TP + t] = f2bf(val);
                    }
                }
            }
        }
}

// ---------------------------------------------------------------------------
// Kernel 4: flash attention. Block = 8 waves, one (b,h), 128 query rows.
// K/V tiles (32 keys) staged per-block in double-buffered LDS with async
// global->LDS copies; per chunk: 8 score WMMAs + online softmax + 8 PV WMMAs.
// ---------------------------------------------------------------------------
__global__ __launch_bounds__(256) void attn_kernel(
    const unsigned short* __restrict__ Qb, const unsigned short* __restrict__ Kb,
    const unsigned short* __restrict__ Vt, unsigned short* __restrict__ AttnB)
{
    __shared__ unsigned short Ks[2][32 * KSTR];      // 2 x 8704 B
    __shared__ unsigned short Vs[2][128 * VSTR];     // 2 x 10240 B
    __shared__ unsigned short Pld[8 * 16 * LDSTR];   // per-wave P strip

    int tid = threadIdx.x, w = tid >> 5, lane = tid & 31;
    int lr = lane & 15, lh = lane >> 4;
    int bh = blockIdx.x >> 3;
    int qb = (blockIdx.x & 7) * 128;
    int b = bh >> 4, h = bh & 15;
    int tbase = qb + w * 16;

    const unsigned short* Qp = Qb + (size_t)bh * TQ * HD;
    const unsigned short* Kp = Kb + (size_t)bh * SEQ * HD;
    const unsigned short* Vp = Vt + (size_t)bh * HD * SEQ;
    unsigned short* Pw = &Pld[w * 16 * LDSTR];

    // stage one 32-key chunk (K tile 32x128, V^T tile 128x32) into LDS buffer
    auto stage = [&](int c, int buf) {
        int s0 = c * 32;
        const unsigned short* Kg = Kp + (size_t)s0 * HD;
        for (int i = 0; i < 2; ++i) {
            int e8 = tid + i * 256;                 // 512 16B chunks
            int row = e8 >> 4, c8 = e8 & 15;        // 16 chunks per 128-elem row
            cp16_g2l(&Ks[buf][row * KSTR + c8 * 8], Kg + (size_t)row * HD + c8 * 8);
        }
        for (int i = 0; i < 2; ++i) {
            int e8 = tid + i * 256;                 // 512 16B chunks
            int row = e8 >> 2, c8 = e8 & 3;         // 4 chunks per 32-elem row
            cp16_g2l(&Vs[buf][row * VSTR + c8 * 8], Vp + (size_t)row * SEQ + s0 + c8 * 8);
        }
    };

    // pin Q fragments (16 rows x 128) in registers: 4 K-fragments
    bf16x16 aq[4];
    {
        const unsigned short* qr = Qp + (size_t)(tbase + lr) * HD;
        for (int f = 0; f < 4; ++f) {
            int kb = f * 32 + lh * 8;
            u16x8 lo = *(const u16x8*)(qr + kb);
            u16x8 hi = *(const u16x8*)(qr + kb + 16);
            aq[f] = mkfrag(lo, hi);
        }
    }

    float rmax[8], rsum[8];
    v8f acco[8];
    const v8f vz = {0.f,0.f,0.f,0.f,0.f,0.f,0.f,0.f};
    for (int j = 0; j < 8; ++j) { rmax[j] = -1e30f; rsum[j] = 0.f; }
    for (int ni = 0; ni < 8; ++ni) acco[ni] = vz;

    const float scale = 0.08838834764831845f;        // 1/sqrt(128)
    int myNch = ((TP + tbase + 15) >> 5) + 1;        // this wave's causal horizon
    int nchB  = ((TP + qb + 127) >> 5) + 1;          // block-uniform chunk count

    stage(0, 0);

    for (int c = 0; c < nchB; ++c) {
        int buf = c & 1;
#if HAVE_ASYNC_LDS
        asm volatile("s_wait_asynccnt 0x0" ::: "memory");
#endif
        __syncthreads();
        if (c + 1 < nchB) stage(c + 1, buf ^ 1);     // overlap fill with compute
        if (c >= myNch) continue;                    // past causal horizon: keep barriers only
        int s0 = c * 32;

        // ---- scores: Q @ K^T  (B col = key, K-dim = hd; from LDS K tile)
        v8f sc[2] = {vz, vz};
        for (int si = 0; si < 2; ++si) {
            const unsigned short* kr = &Ks[buf][(si * 16 + lr) * KSTR];
            for (int f = 0; f < 4; ++f) {
                int kb = f * 32 + lh * 8;
                u16x8 lo = *(const u16x8*)(kr + kb);
                u16x8 hi = *(const u16x8*)(kr + kb + 16);
                sc[si] = wmma_bf16(aq[f], mkfrag(lo, hi), sc[si]);
            }
        }
        // ---- scale + causal mask (query global pos TP+t attends s <= pos)
        for (int si = 0; si < 2; ++si)
            for (int j = 0; j < 8; ++j) {
                int qpos = TP + tbase + lh * 8 + j;
                int s = s0 + si * 16 + lr;
                float v = sc[si][j] * scale;
                sc[si][j] = (s <= qpos) ? v : -1e30f;
            }
        // ---- online softmax (row stats replicated across 16-lane halves)
        float nm[8], sOld[8];
        for (int j = 0; j < 8; ++j) {
            float cm = fmaxf(sc[0][j], sc[1][j]);
            cm = fmaxf(cm, __shfl_xor(cm, 1));
            cm = fmaxf(cm, __shfl_xor(cm, 2));
            cm = fmaxf(cm, __shfl_xor(cm, 4));
            cm = fmaxf(cm, __shfl_xor(cm, 8));
            nm[j] = fmaxf(rmax[j], cm);
            sOld[j] = __expf(rmax[j] - nm[j]);
            rmax[j] = nm[j];
        }
        for (int si = 0; si < 2; ++si)
            for (int j = 0; j < 8; ++j) sc[si][j] = __expf(sc[si][j] - nm[j]);
        for (int j = 0; j < 8; ++j) {
            float cs = sc[0][j] + sc[1][j];
            cs += __shfl_xor(cs, 1);
            cs += __shfl_xor(cs, 2);
            cs += __shfl_xor(cs, 4);
            cs += __shfl_xor(cs, 8);
            rsum[j] = rsum[j] * sOld[j] + cs;
        }
        for (int ni = 0; ni < 8; ++ni)
            for (int j = 0; j < 8; ++j) acco[ni][j] *= sOld[j];

        // ---- transpose P (C layout -> A layout) through per-wave LDS strip
        for (int si = 0; si < 2; ++si)
            for (int j = 0; j < 8; ++j)
                Pw[(lh * 8 + j) * LDSTR + si * 16 + lr] = f2bf(sc[si][j]);
        asm volatile("s_wait_dscnt 0x0" ::: "memory");
        bf16x16 pf;
        {
            int kb = lh * 8;
            u16x8 lo = *(const u16x8*)(&Pw[lr * LDSTR + kb]);
            u16x8 hi = *(const u16x8*)(&Pw[lr * LDSTR + kb + 16]);
            pf = mkfrag(lo, hi);
        }
        // ---- P @ V  (B col = hd, K-dim = s; from LDS V^T tile)
        for (int ni = 0; ni < 8; ++ni) {
            const unsigned short* vr = &Vs[buf][(ni * 16 + lr) * VSTR + lh * 8];
            u16x8 lo = *(const u16x8*)(vr);
            u16x8 hi = *(const u16x8*)(vr + 16);
            acco[ni] = wmma_bf16(pf, mkfrag(lo, hi), acco[ni]);
        }
    }

    // ---- normalize and emit bf16 attention output [B*TQ, D]
    float inv[8];
    for (int j = 0; j < 8; ++j) inv[j] = 1.0f / rsum[j];
    for (int ni = 0; ni < 8; ++ni)
        for (int j = 0; j < 8; ++j) {
            int t = tbase + lh * 8 + j;
            int col = h * HD + ni * 16 + lr;
            AttnB[((size_t)b * TQ + t) * D_MODEL + col] = f2bf(acco[ni][j] * inv[j]);
        }
}

// ---------------------------------------------------------------------------
// host launcher
// ---------------------------------------------------------------------------
extern "C" void kernel_launch(void* const* d_in, const int* in_sizes, int n_in,
                              void* d_out, int out_size, void* d_ws, size_t ws_size,
                              hipStream_t stream) {
    (void)in_sizes; (void)n_in; (void)out_size; (void)ws_size;
    const float* x  = (const float*)d_in[0];
    const float* kc = (const float*)d_in[1];
    const float* vc = (const float*)d_in[2];
    const float* Wq = (const float*)d_in[3];
    const float* bq = (const float*)d_in[4];
    const float* Wk = (const float*)d_in[5];
    const float* bk = (const float*)d_in[6];
    const float* Wv = (const float*)d_in[7];
    const float* bv = (const float*)d_in[8];
    const float* Wo = (const float*)d_in[9];
    const float* bo = (const float*)d_in[10];

    float* out  = (float*)d_out;                  // [4,1024,2048]
    float* Kout = out + 8388608ull;               // [4,16,4096,128]
    float* Vout = out + 41943040ull;              // [4,16,4096,128]

    // workspace (bf16 as u16): Qb 16MB | Kb 64MB | Vt 64MB | AttnB 16MB
    unsigned short* ws = (unsigned short*)d_ws;
    unsigned short* Qb = ws;
    unsigned short* Kb = ws + 8388608ull;
    unsigned short* Vt = ws + 41943040ull;
    unsigned short* At = ws + 75497472ull;

    copy_caches<<<24576, 256, 0, stream>>>(kc, vc, Kout, Vout, Kb);
    vt_cache<<<6144, 256, 0, stream>>>(vc, Vt);
    proj_gemm<0><<<512, 256, 0, stream>>>(x, nullptr, Wq, bq, nullptr, Qb);
    proj_gemm<1><<<512, 256, 0, stream>>>(x, nullptr, Wk, bk, Kout, Kb);
    proj_gemm<2><<<512, 256, 0, stream>>>(x, nullptr, Wv, bv, Vout, Vt);
    attn_kernel<<<512, 256, 0, stream>>>(Qb, Kb, Vt, At);
    proj_gemm<3><<<512, 256, 0, stream>>>(nullptr, At, Wo, bo, out, nullptr);
}
